// DeformableTransformer_41446434407088
// MI455X (gfx1250) — matrix operand order repaired and checked by reading, not compiled, not run on hardware
//
#include <hip/hip_runtime.h>
#include <cstddef>

// ---------------- problem constants ----------------
#define D_    256
#define DFFN_ 1024
#define NH_   8
#define DH_   32
#define LL_   4
#define PP_   4
#define NLE_  6
#define NLD_  6
#define B_    2
#define Q_    300
#define S_    8160   // 64*96 + 32*48 + 16*24 + 8*12

__constant__ int cH[4]  = {64, 32, 16, 8};
__constant__ int cW[4]  = {96, 48, 24, 12};
__constant__ int cLS[4] = {0, 6144, 7680, 8064};

// ---------------- vector types ----------------
typedef __attribute__((ext_vector_type(16))) __bf16         v16bf;
typedef __attribute__((ext_vector_type(2)))  __bf16         v2bf;
typedef __attribute__((ext_vector_type(8)))  float          v8f;
typedef __attribute__((ext_vector_type(4)))  float          f4;
typedef __attribute__((ext_vector_type(4)))  unsigned int   v4u;
typedef __attribute__((ext_vector_type(2)))  unsigned int   v2u;

union FragU { v4u q[2]; v16bf bf; };

__device__ __forceinline__ unsigned short f32_to_bf16u(float f) {
  union { float f; unsigned u; } x; x.f = f;
  unsigned r = x.u + 0x7FFFu + ((x.u >> 16) & 1u);   // round-to-nearest-even
  return (unsigned short)(r >> 16);
}
__device__ __forceinline__ unsigned pk2(float lo, float hi) {
#if __has_builtin(__builtin_amdgcn_cvt_pk_bf16_f32)
  v2bf r = __builtin_amdgcn_cvt_pk_bf16_f32(lo, hi);
  return __builtin_bit_cast(unsigned, r);
#else
  return (unsigned)f32_to_bf16u(lo) | ((unsigned)f32_to_bf16u(hi) << 16);
#endif
}

// ---------------- weight pre-pack: Wp[n][k] = bf16(W[k][wcol0+n]) ----------------
// Packs a K x N weight slice into bf16 fragment order (row = output col, 16B runs of k).
__global__ void __launch_bounds__(256)
pack_w_kernel(const float* __restrict__ W, int ldw, int wcol0,
              unsigned short* __restrict__ Wp, int N, int K)
{
  int i = blockIdx.x * blockDim.x + threadIdx.x;
  int total = N * (K >> 3);
  if (i >= total) return;
  int n  = i % N;            // consecutive threads -> consecutive n (coalesced reads)
  int k8 = i / N;            // block of 8 k values
  const float* src = W + (size_t)(k8 * 8) * ldw + wcol0 + n;
  v4u o;
#pragma unroll
  for (int j = 0; j < 4; ++j)
    o[j] = pk2(src[(size_t)(2 * j) * ldw], src[(size_t)(2 * j + 1) * ldw]);
  *(v4u*)(Wp + (size_t)n * K + k8 * 8) = o;
}

// ---------------- generic WMMA GEMM ----------------
// C[M,N] = act( (A (+A2)) [M,K] @ Wp^T + bias )   (ldC = N, ldA = K; Wp is [N][K] bf16)
// Requires: K % 32 == 0, N % 128 == 0.
#define BM 64
#define BN 128
#define BK 32

template <bool HAS_A2, bool RELU>
__global__ void __launch_bounds__(256)
wmma_gemm_kernel(const float* __restrict__ A, const float* __restrict__ A2,
                 const unsigned short* __restrict__ Wp,
                 const float* __restrict__ bias,
                 float* __restrict__ C, int M, int N, int K)
{
  // ping-pong A staging buffer (B comes straight from pre-packed global bf16)
  __shared__ __align__(16) unsigned short As[2][BM][BK + 8];

  const int t    = threadIdx.x;
  const int lane = t & 31;
  const int w    = t >> 5;        // 8 waves
  const int wm   = w >> 2;        // M sub-tile (32 rows)
  const int wn   = w & 3;         // N sub-tile (32 cols)
  const int bm   = blockIdx.y * BM;
  const int bn   = blockIdx.x * BN;
  const int hh   = lane >> 4;     // half-wave
  const int nn   = lane & 15;

  // A staging: thread covers rows (ar0, ar0+32), 4 consecutive k each
  const int ar0 = t >> 3;
  const int ac  = (t & 7) * 4;

  const int nt = K / BK;

  v8f acc[2][2] = {{{}, {}}, {{}, {}}};
  f4 a0, a1;

  auto ldTile = [&](int kt) {
    const int kbase = kt * BK;
    a0 = (f4){0.f, 0.f, 0.f, 0.f};
    a1 = (f4){0.f, 0.f, 0.f, 0.f};
    const int r0 = bm + ar0, r1 = bm + ar0 + 32;
    if (r0 < M) {
      a0 = *(const f4*)(A + (size_t)r0 * K + kbase + ac);
      if (HAS_A2) a0 += *(const f4*)(A2 + (size_t)r0 * K + kbase + ac);
    }
    if (r1 < M) {
      a1 = *(const f4*)(A + (size_t)r1 * K + kbase + ac);
      if (HAS_A2) a1 += *(const f4*)(A2 + (size_t)r1 * K + kbase + ac);
    }
  };
  auto stTile = [&](int buf) {
    v2u s0 = {pk2(a0[0], a0[1]), pk2(a0[2], a0[3])};
    *(v2u*)&As[buf][ar0][ac] = s0;
    v2u s1 = {pk2(a1[0], a1[1]), pk2(a1[2], a1[3])};
    *(v2u*)&As[buf][ar0 + 32][ac] = s1;
  };

  // per-wave B fragment base: row (bn + wn*32 + nn) of Wp, 16 consecutive k per half
  const unsigned short* wrow0 = Wp + (size_t)(bn + wn * 32 + nn) * K;
  const unsigned short* wrow1 = wrow0 + (size_t)16 * K;

  ldTile(0);
  stTile(0);
  __syncthreads();

  for (int kt = 0; kt < nt; ++kt) {
    const int cur = kt & 1;
    const bool more = (kt + 1) < nt;
    if (more) ldTile(kt + 1);                       // overlap HBM with LDS+WMMA
    if (kt + 2 < nt && (bm + ar0) < M)              // global_prefetch_b8 for k+2
      __builtin_prefetch(A + (size_t)(bm + ar0) * K + (kt + 2) * BK + ac, 0, 1);

    // B fragments: direct global b128 loads from packed bf16 (L2 resident)
    FragU fb[2];
    const unsigned short* wp0 = wrow0 + kt * BK + hh * 16;
    fb[0].q[0] = *(const v4u*)(wp0);
    fb[0].q[1] = *(const v4u*)(wp0 + 8);
    const unsigned short* wp1 = wrow1 + kt * BK + hh * 16;
    fb[1].q[0] = *(const v4u*)(wp1);
    fb[1].q[1] = *(const v4u*)(wp1 + 8);

    // A fragments: lane = row; 16 consecutive k per half
    FragU fa[2];
#pragma unroll
    for (int mi = 0; mi < 2; ++mi) {
      const unsigned short* ap = &As[cur][wm * 32 + mi * 16 + nn][0];
      fa[mi].q[0] = *(const v4u*)(ap + hh * 16);
      fa[mi].q[1] = *(const v4u*)(ap + hh * 16 + 8);
    }

#pragma unroll
    for (int mi = 0; mi < 2; ++mi)
#pragma unroll
      for (int ni = 0; ni < 2; ++ni)
        acc[mi][ni] = __builtin_amdgcn_wmma_f32_16x16x32_bf16(
            false, fa[mi].bf, false, fb[ni].bf, (short)0, acc[mi][ni], false, false);

    if (more) stTile(cur ^ 1);
    __syncthreads();
  }

  // C/D layout: VGPR r -> M = r + 8*half, lane -> N = lane%16
#pragma unroll
  for (int mi = 0; mi < 2; ++mi) {
#pragma unroll
    for (int ni = 0; ni < 2; ++ni) {
      const int col = bn + wn * 32 + ni * 16 + nn;
#pragma unroll
      for (int r = 0; r < 8; ++r) {
        const int row = bm + wm * 32 + mi * 16 + hh * 8 + r;
        if (row < M) {
          float v0 = acc[mi][ni][r] + bias[col];
          if (RELU) v0 = fmaxf(v0, 0.f);
          C[(size_t)row * N + col] = v0;
        }
      }
    }
  }
}

// ---------------- fused residual-add + LayerNorm (blockDim = D_ = 256) ----------------
__global__ void __launch_bounds__(256)
add_layernorm_kernel(const float* __restrict__ a, const float* __restrict__ b,
                     float* __restrict__ out,
                     const float* __restrict__ g, const float* __restrict__ beta)
{
  int row = blockIdx.x;
  int t = threadIdx.x;
  float v = a[(size_t)row * D_ + t] + b[(size_t)row * D_ + t];

  __shared__ float red[16];
  float s = v, s2 = v * v;
#pragma unroll
  for (int o = 16; o; o >>= 1) { s += __shfl_xor(s, o, 32); s2 += __shfl_xor(s2, o, 32); }
  if ((t & 31) == 0) { red[t >> 5] = s; red[8 + (t >> 5)] = s2; }
  __syncthreads();
  float ts = 0.f, ts2 = 0.f;
#pragma unroll
  for (int i = 0; i < 8; ++i) { ts += red[i]; ts2 += red[8 + i]; }
  float mean = ts * (1.f / D_);
  float var  = ts2 * (1.f / D_) - mean * mean;
  out[(size_t)row * D_ + t] = (v - mean) * rsqrtf(var + 1e-5f) * g[t] + beta[t];
}

// ---------------- wave-per-row softmax ----------------
__global__ void __launch_bounds__(256)
softmax_rows_kernel(float* __restrict__ x, int rows, int n)
{
  int wid  = (blockIdx.x * blockDim.x + threadIdx.x) >> 5;
  int lane = threadIdx.x & 31;
  if (wid >= rows) return;
  float* row = x + (size_t)wid * n;
  float m = -3.4e38f;
  for (int k = lane; k < n; k += 32) m = fmaxf(m, row[k]);
#pragma unroll
  for (int o = 16; o; o >>= 1) m = fmaxf(m, __shfl_xor(m, o, 32));
  float s = 0.f;
  for (int k = lane; k < n; k += 32) s += __expf(row[k] - m);
#pragma unroll
  for (int o = 16; o; o >>= 1) s += __shfl_xor(s, o, 32);
  float inv = 1.f / s;
  for (int k = lane; k < n; k += 32) row[k] = __expf(row[k] - m) * inv;
}

// ---------------- multi-scale deformable sampling: wave per (b,q,h), lane = channel ----------------
__global__ void __launch_bounds__(256)
deform_attn_kernel(const float* __restrict__ val, const float* __restrict__ off,
                   const float* __restrict__ attw, const float* __restrict__ refq,
                   float* __restrict__ outp, int Qn)
{
  int wid  = (blockIdx.x * blockDim.x + threadIdx.x) >> 5;
  int lane = threadIdx.x & 31;                 // channel d
  int total = B_ * Qn * NH_;
  if (wid >= total) return;
  int h = wid % NH_;
  int q = (wid / NH_) % Qn;
  int b = wid / (NH_ * Qn);

  float rx, ry;
  if (refq) {                    // decoder refs
    rx = refq[q * 2 + 0]; ry = refq[q * 2 + 1];
  } else {                       // encoder: analytic grid ref from flattened s
    int s = q;
    int l = 3;
    if (s < 6144) l = 0; else if (s < 7680) l = 1; else if (s < 8064) l = 2;
    int Hl = cH[l], Wl = cW[l], o = s - cLS[l];
    int iy = o / Wl, ix = o - iy * Wl;
    rx = (ix + 0.5f) / (float)Wl;
    ry = (iy + 0.5f) / (float)Hl;
  }

  const float* offb = off  + ((size_t)(b * Qn + q) * NH_ + h) * (LL_ * PP_ * 2);
  const float* aw   = attw + ((size_t)(b * Qn + q) * NH_ + h) * (LL_ * PP_);

  float acc = 0.f;
#pragma unroll
  for (int l = 0; l < LL_; ++l) {
    int Hl = cH[l], Wl = cW[l], base = cLS[l];
#pragma unroll
    for (int p = 0; p < PP_; ++p) {
      float ox = offb[(l * PP_ + p) * 2 + 0];
      float oy = offb[(l * PP_ + p) * 2 + 1];
      float wgt = aw[l * PP_ + p];
      float x = rx * Wl + ox - 0.5f;   // (ref + off/[W,H])*W - 0.5
      float y = ry * Hl + oy - 0.5f;
      float x0 = floorf(x), y0 = floorf(y);
      float sample = 0.f;
#pragma unroll
      for (int dy = 0; dy < 2; ++dy) {
#pragma unroll
        for (int dx = 0; dx < 2; ++dx) {
          float xi = x0 + dx, yi = y0 + dy;
          float bw = (1.f - fabsf(x - xi)) * (1.f - fabsf(y - yi));
          bool valid = (xi >= 0.f) && (xi < (float)Wl) && (yi >= 0.f) && (yi < (float)Hl);
          int cx = (int)fminf(fmaxf(xi, 0.f), (float)(Wl - 1));
          int cy = (int)fminf(fmaxf(yi, 0.f), (float)(Hl - 1));
          int sidx = base + cy * Wl + cx;
          float v = val[((size_t)(b * S_ + sidx)) * D_ + h * DH_ + lane];
          sample += v * (valid ? bw : 0.f);
        }
      }
      acc += sample * wgt;
    }
  }
  outp[((size_t)(b * Qn + q)) * D_ + h * DH_ + lane] = acc;
}

// ---------------- decoder init: broadcast query_embed into qpos / tgt ----------------
__global__ void dec_init_kernel(const float* __restrict__ qe,
                                float* __restrict__ qpos, float* __restrict__ xd)
{
  int i = blockIdx.x * blockDim.x + threadIdx.x;
  if (i >= B_ * Q_ * D_) return;
  int t = i % D_;
  int q = (i / D_) % Q_;
  qpos[i] = qe[(size_t)q * (2 * D_) + t];
  xd[i]   = qe[(size_t)q * (2 * D_) + D_ + t];
}

// ---------------- decoder reference points: sigmoid(qe[:, :D] @ ref_w + ref_b) ----------------
__global__ void dec_ref_kernel(const float* __restrict__ qe, const float* __restrict__ rw,
                               const float* __restrict__ rb, float* __restrict__ refq)
{
  int i = blockIdx.x * blockDim.x + threadIdx.x;
  if (i >= Q_ * 2) return;
  int j = i & 1, q = i >> 1;
  float acc = rb[j];
  for (int k = 0; k < D_; ++k) acc += qe[(size_t)q * (2 * D_) + k] * rw[k * 2 + j];
  refq[i] = 1.f / (1.f + __expf(-acc));
}

// ---------------- decoder self-attn scores ----------------
__global__ void self_att_scores_kernel(const float* __restrict__ qk, float* __restrict__ att)
{
  int i = blockIdx.x * blockDim.x + threadIdx.x;
  if (i >= B_ * NH_ * Q_ * Q_) return;
  int k  = i % Q_;
  int qq = (i / Q_) % Q_;
  int h  = (i / (Q_ * Q_)) % NH_;
  int b  = i / (Q_ * Q_ * NH_);
  const float* qp = qk + ((size_t)(b * Q_ + qq)) * 512 + h * DH_;
  const float* kp = qk + ((size_t)(b * Q_ + k)) * 512 + 256 + h * DH_;
  float acc = 0.f;
#pragma unroll
  for (int d = 0; d < DH_; ++d) acc += qp[d] * kp[d];
  att[i] = acc * 0.17677669529663687f;  // 1/sqrt(32)
}

// ---------------- decoder self-attn apply ----------------
__global__ void self_att_apply_kernel(const float* __restrict__ att,
                                      const float* __restrict__ vv, float* __restrict__ sa)
{
  int i = blockIdx.x * blockDim.x + threadIdx.x;
  if (i >= B_ * Q_ * D_) return;
  int d = i % DH_;
  int h = (i / DH_) % NH_;
  int q = (i / D_) % Q_;
  int b = i / (D_ * Q_);
  const float* ar = att + (((size_t)(b * NH_ + h) * Q_) + q) * Q_;
  const float* vr = vv + (size_t)b * Q_ * D_ + h * DH_ + d;
  float acc = 0.f;
  for (int k = 0; k < Q_; ++k) acc += ar[k] * vr[(size_t)k * D_];
  sa[i] = acc;
}

// ---------------- host orchestration ----------------
extern "C" void kernel_launch(void* const* d_in, const int* in_sizes, int n_in,
                              void* d_out, int out_size, void* d_ws, size_t ws_size,
                              hipStream_t stream)
{
  const float* src  = (const float*)d_in[0];
  const float* pos  = (const float*)d_in[1];
  const float* qe   = (const float*)d_in[2];
  const float* refw = (const float*)d_in[3];
  const float* refb = (const float*)d_in[4];

  auto EP = [&](int i, int l) -> const float* {
    return (const float*)d_in[i] + (size_t)l * ((size_t)in_sizes[i] / NLE_);
  };
  auto DP = [&](int i, int l) -> const float* {
    return (const float*)d_in[i] + (size_t)l * ((size_t)in_sizes[i] / NLD_);
  };
  // enc: 5 off_w 6 off_b 7 attw_w 8 attw_b 9 val_w 10 val_b 11 out_w 12 out_b
  //      13 ln1_g 14 ln1_b 15 f1_w 16 f1_b 17 f2_w 18 f2_b 19 ln2_g 20 ln2_b
  // dec: 21 off_w 22 off_b 23 attw_w 24 attw_b 25 val_w 26 val_b 27 out_w 28 out_b
  //      29 qkv_w 30 qkv_b 31 oa_w 32 oa_b 33 ln1_g 34 ln1_b 35 ln2_g 36 ln2_b
  //      37 f1_w 38 f1_b 39 f2_w 40 f2_b 41 ln3_g 42 ln3_b

  const int BS = B_ * S_;   // 16320 rows
  const int BQ = B_ * Q_;   // 600 rows

  float* ws = (float*)d_ws;
  size_t o = 0;
  auto alloc = [&](size_t n) { float* p = ws + o; o += (n + 63) & ~(size_t)63; return p; };
  float* x    = alloc((size_t)BS * D_);
  float* val  = alloc((size_t)BS * D_);
  float* offb = alloc((size_t)BS * 256);
  float* attw = alloc((size_t)BS * 128);
  float* samp = alloc((size_t)BS * D_);
  float* proj = alloc((size_t)BS * D_);
  float* ffn  = alloc((size_t)BS * DFFN_);
  float* xd   = alloc((size_t)BQ * D_);
  float* qpos = alloc((size_t)BQ * D_);
  float* refq = alloc((size_t)Q_ * 2);
  float* qk   = alloc((size_t)BQ * 512);
  float* vv   = alloc((size_t)BQ * D_);
  float* att  = alloc((size_t)B_ * NH_ * Q_ * Q_);
  float* sa   = alloc((size_t)BQ * D_);
  unsigned short* wpack = (unsigned short*)alloc((size_t)DFFN_ * D_ / 2); // 256K bf16 max
  (void)ws_size; (void)n_in; (void)out_size;

  // pack weights then run WMMA GEMM (in-order stream => single wpack buffer is safe)
  auto gemm = [&](const float* A, const float* A2, const float* W, int ldw, int wcol0,
                  const float* bias, float* C, int M, int N, int K, int act) {
    int total = N * (K >> 3);
    pack_w_kernel<<<(total + 255) / 256, 256, 0, stream>>>(W, ldw, wcol0, wpack, N, K);
    dim3 g(N / BN, (M + BM - 1) / BM);
    if (A2) {
      if (act) wmma_gemm_kernel<true, true><<<g, 256, 0, stream>>>(A, A2, wpack, bias, C, M, N, K);
      else     wmma_gemm_kernel<true, false><<<g, 256, 0, stream>>>(A, A2, wpack, bias, C, M, N, K);
    } else {
      if (act) wmma_gemm_kernel<false, true><<<g, 256, 0, stream>>>(A, nullptr, wpack, bias, C, M, N, K);
      else     wmma_gemm_kernel<false, false><<<g, 256, 0, stream>>>(A, nullptr, wpack, bias, C, M, N, K);
    }
  };
  auto add_ln = [&](const float* a, const float* b, float* out,
                    const float* g, const float* beta, int rows) {
    add_layernorm_kernel<<<rows, 256, 0, stream>>>(a, b, out, g, beta);
  };
  auto softmax = [&](float* p, int rows, int n) {
    softmax_rows_kernel<<<(rows + 7) / 8, 256, 0, stream>>>(p, rows, n);
  };

  // x = src_flatten (inputs stay immutable)
  hipMemcpyAsync(x, src, sizeof(float) * (size_t)BS * D_, hipMemcpyDeviceToDevice, stream);

  // ======================= encoder =======================
  for (int l = 0; l < NLE_; ++l) {
    gemm(x, pos,     EP(5, l),  256, 0, EP(6, l),  offb, BS, 256, 256, 0);
    gemm(x, pos,     EP(7, l),  128, 0, EP(8, l),  attw, BS, 128, 256, 0);
    gemm(x, nullptr, EP(9, l),  256, 0, EP(10, l), val,  BS, 256, 256, 0);
    softmax(attw, BS * NH_, LL_ * PP_);
    {
      int waves = B_ * S_ * NH_;
      deform_attn_kernel<<<(waves + 7) / 8, 256, 0, stream>>>(val, offb, attw, nullptr, samp, S_);
    }
    gemm(samp, nullptr, EP(11, l), 256, 0, EP(12, l), proj, BS, 256, 256, 0);
    add_ln(x, proj, x, EP(13, l), EP(14, l), BS);                              // ln1
    gemm(x, nullptr,   EP(15, l), 1024, 0, EP(16, l), ffn,  BS, 1024, 256, 1); // FFN up + relu
    gemm(ffn, nullptr, EP(17, l), 256,  0, EP(18, l), proj, BS, 256, 1024, 0); // FFN down
    add_ln(x, proj, x, EP(19, l), EP(20, l), BS);                              // ln2
  }
  // x is now `memory`

  // ======================= decoder init =======================
  dec_init_kernel<<<(BQ * D_ + 255) / 256, 256, 0, stream>>>(qe, qpos, xd);
  dec_ref_kernel<<<(Q_ * 2 + 255) / 256, 256, 0, stream>>>(qe, refw, refb, refq);

  // ======================= decoder =======================
  for (int l = 0; l < NLD_; ++l) {
    // --- self attention ---
    gemm(xd, qpos,    DP(29, l), 768, 0,   DP(30, l),       qk, BQ, 512, 256, 0); // q|k
    gemm(xd, nullptr, DP(29, l), 768, 512, DP(30, l) + 512, vv, BQ, 256, 256, 0); // v
    self_att_scores_kernel<<<(B_ * NH_ * Q_ * Q_ + 255) / 256, 256, 0, stream>>>(qk, att);
    softmax(att, B_ * NH_ * Q_, Q_);
    self_att_apply_kernel<<<(BQ * D_ + 255) / 256, 256, 0, stream>>>(att, vv, sa);
    gemm(sa, nullptr, DP(31, l), 256, 0, DP(32, l), proj, BQ, 256, 256, 0);       // oa
    add_ln(xd, proj, xd, DP(35, l), DP(36, l), BQ);                               // ln2

    // --- cross deformable attention over memory ---
    gemm(xd, qpos,    DP(21, l), 256, 0, DP(22, l), offb, BQ, 256, 256, 0);
    gemm(xd, qpos,    DP(23, l), 128, 0, DP(24, l), attw, BQ, 128, 256, 0);
    gemm(x,  nullptr, DP(25, l), 256, 0, DP(26, l), val,  BS, 256, 256, 0);
    softmax(attw, BQ * NH_, LL_ * PP_);
    {
      int waves = B_ * Q_ * NH_;
      deform_attn_kernel<<<(waves + 7) / 8, 256, 0, stream>>>(val, offb, attw, refq, samp, Q_);
    }
    gemm(samp, nullptr, DP(27, l), 256, 0, DP(28, l), proj, BQ, 256, 256, 0);
    add_ln(xd, proj, xd, DP(33, l), DP(34, l), BQ);                               // ln1

    // --- FFN ---
    gemm(xd, nullptr,  DP(37, l), 1024, 0, DP(38, l), ffn,  BQ, 1024, 256, 1);
    gemm(ffn, nullptr, DP(39, l), 256,  0, DP(40, l), proj, BQ, 256, 1024, 0);
    add_ln(xd, proj, xd, DP(41, l), DP(42, l), BQ);                               // ln3
  }

  hipMemcpyAsync(d_out, xd, sizeof(float) * (size_t)BQ * D_, hipMemcpyDeviceToDevice, stream);
}
